// MemoryLayer_12764642804204
// MI455X (gfx1250) — matrix operand order repaired and checked by reading, not compiled, not run on hardware
//
#include <hip/hip_runtime.h>
#include <hip/hip_bf16.h>

typedef __attribute__((ext_vector_type(16))) _Float16 v16h;
typedef __attribute__((ext_vector_type(8)))  float    v8f;

#define B_ 8
#define T_ 2048
#define M_ 8
#define D_ 64
#define R_ 128
#define O_ 32

// ---- WMMA helpers ---------------------------------------------------------

__device__ inline v8f wmma_f16(v16h a, v16h b, v8f c) {
  // D = A(16x32 f16) * B(32x16 f16) + C(16x16 f32)
  return __builtin_amdgcn_wmma_f32_16x16x32_f16(
      /*neg_a=*/false, a, /*neg_b=*/false, b,
      /*c_mod=*/(short)0, c, /*reuse_a=*/false, /*reuse_b=*/false);
}

// A fragment (16x32, f16): row = lane%16; lane-half h selects K sub-blocks.
__device__ inline v16h load_a_frag(const _Float16* p, int stride, int lane, int k0) {
  int row = lane & 15, h = lane >> 4;
  const _Float16* r = p + row * stride + k0;
  v16h a;
#pragma unroll
  for (int v = 0; v < 4; ++v) {
    a[2 * v]     = r[h * 8 + 2 * v];
    a[2 * v + 1] = r[h * 8 + 2 * v + 1];
    a[8 + 2 * v]     = r[16 + h * 8 + 2 * v];
    a[8 + 2 * v + 1] = r[16 + h * 8 + 2 * v + 1];
  }
  return a;
}

// B fragment (32x16, f16) from an n-major (transposed) [N][K] array:
// col n = n0 + lane%16; elem i<8 -> K = k0 + h*8 + i, i>=8 -> K = k0 + 16 + h*8 + (i-8)
__device__ inline v16h load_b_frag(const _Float16* bt, int stride, int lane, int n0, int k0) {
  int n = n0 + (lane & 15), h = lane >> 4;
  const _Float16* r = bt + n * stride + k0 + h * 8;
  v16h b;
#pragma unroll
  for (int i = 0; i < 16; ++i) b[i] = r[(i < 8) ? i : (i + 8)];
  return b;
}

// ---- CDNA5 async global<->LDS DMA (VGLOBAL async encoding) ---------------

template <typename T>
__device__ inline unsigned lds_addr(const T* p) {
  // generic LDS pointer: addr[31:0] is the LDS byte offset (flat aperture map)
  return (unsigned)(unsigned long long)(const void*)p;
}

__device__ inline void async_load_b128(unsigned lds_off, const void* gptr) {
  asm volatile("global_load_async_to_lds_b128 %0, %1, off"
               :: "v"(lds_off), "v"(gptr) : "memory");
}

__device__ inline void async_store_b128(const void* gptr, unsigned lds_off) {
  asm volatile("global_store_async_from_lds_b128 %0, %1, off"
               :: "v"(gptr), "v"(lds_off) : "memory");
}

__device__ inline void wait_async0() {
  asm volatile("s_wait_asynccnt 0" ::: "memory");
}

// ---- branch-free fast tanh (Pade 7,6), keeps VALU co-executable -----------

__device__ inline float fast_tanh(float x) {
  x = fminf(fmaxf(x, -4.97f), 4.97f);
  float x2 = x * x;
  float num = fmaf(x2, fmaf(x2, fmaf(x2, 1.0f, 378.0f), 17325.0f), 135135.0f) * x;
  float den = fmaf(x2, fmaf(x2, fmaf(x2, 28.0f, 3150.0f), 62370.0f), 135135.0f);
  return __fdividef(num, den);
}

// ---- kernel 0: weight transpose + f32/complex64 -> f16 --------------------

__global__ __launch_bounds__(256) void k0_prep(
    const float* __restrict__ proj, const float2* __restrict__ Win,
    const float2* __restrict__ Wout,
    _Float16* __restrict__ projT, _Float16* __restrict__ winReT,
    _Float16* __restrict__ winImT, _Float16* __restrict__ woutReT,
    _Float16* __restrict__ woutImNegT)
{
  int i = blockIdx.x * 256 + threadIdx.x;           // covers up to M*D*R = 65536
  if (i < M_ * D_ * D_) {                           // projT[m][e][d] = proj[m][d][e]
    int m = i / (D_ * D_), rem = i % (D_ * D_);
    int e = rem / D_, d = rem % D_;
    projT[i] = (_Float16)proj[(m * D_ + d) * D_ + e];
  }
  if (i < M_ * D_ * R_) {                           // winT[m][r][d] = Win[m][d][r]
    int m = i / (D_ * R_), rem = i % (D_ * R_);
    int r = rem / D_, d = rem % D_;
    float2 v = Win[((size_t)m * D_ + d) * R_ + r];
    winReT[i] = (_Float16)v.x;
    winImT[i] = (_Float16)v.y;
  }
  if (i < M_ * O_ * R_) {                           // woutT[m][o][r] = Wout[m][r][o]
    int m = i / (O_ * R_), rem = i % (O_ * R_);
    int o = rem / R_, r = rem % R_;
    float2 v = Wout[((size_t)m * R_ + r) * O_ + o];
    woutReT[i]    = (_Float16)v.x;
    woutImNegT[i] = (_Float16)(-v.y);               // fold "- Im" into the weight
  }
}

// ---- kernel 1: X_proj = tanh(X @ proj) ; lr = softmax_m(logits) -----------
// grid: B * T/16 blocks, 256 threads = 8 waves, wave m <-> unit m.

__global__ __launch_bounds__(256) void k1_proj_lr(
    const float* __restrict__ X, const _Float16* __restrict__ projT,
    const float* __restrict__ alr, const float* __restrict__ temp,
    _Float16* __restrict__ xp_ws, float* __restrict__ lr_ws)
{
  __shared__ _Float16 xTile[16 * D_];         // 2 KB
  __shared__ _Float16 xpAll[M_][16 * D_];     // 16 KB
  __shared__ float    logits[16 * M_];

  int tid = threadIdx.x;
  int b  = blockIdx.x / (T_ / 16);
  int t0 = (blockIdx.x % (T_ / 16)) * 16;

  for (int i = tid; i < 16 * D_; i += 256)
    xTile[i] = (_Float16)X[((size_t)(b * T_ + t0 + (i >> 6))) * D_ + (i & 63)];
  __syncthreads();

  int w = tid >> 5, lane = tid & 31;
  int m = w;
  int h = lane >> 4, nl = lane & 15;
  const _Float16* pB = projT + m * D_ * D_;

  v16h a0 = load_a_frag(xTile, D_, lane, 0);
  v16h a1 = load_a_frag(xTile, D_, lane, 32);

#pragma unroll
  for (int nt = 0; nt < 4; ++nt) {
    v8f acc = {};
    acc = wmma_f16(a0, load_b_frag(pB, D_, lane, nt * 16, 0), acc);
    acc = wmma_f16(a1, load_b_frag(pB, D_, lane, nt * 16, 32), acc);
#pragma unroll
    for (int g = 0; g < 8; ++g) {
      int t = g + 8 * h;                       // C/D layout: row = g + 8*(lane>=16)
      int d = nt * 16 + nl;
      xpAll[m][t * D_ + d] = (_Float16)fast_tanh(acc[g]);
    }
  }
  __syncthreads();

  // DMA X_proj tile LDS -> global (16B per lane, ASYNCcnt-tracked)
  for (int i = tid; i < M_ * 16 * 8; i += 256) {   // 1024 x 16B segments
    int row = i >> 3, seg = i & 7;
    int mm = row >> 4, t = row & 15;
    const _Float16* g = xp_ws + ((size_t)(b * T_ + t0 + t) * M_ + mm) * D_ + seg * 8;
    async_store_b128(g, lds_addr(&xpAll[mm][t * D_ + seg * 8]));
  }

  if (lane < 16) {                             // logits[t][m]
    float s = 0.f;
    for (int d = 0; d < D_; ++d)
      s += (float)xpAll[m][lane * D_ + d] * alr[m * D_ + d];
    logits[lane * M_ + m] = s / temp[m];
  }
  __syncthreads();

  if (lane < 16) {                             // softmax over units
    float mx = -1e30f;
    for (int j = 0; j < M_; ++j) mx = fmaxf(mx, logits[lane * M_ + j]);
    float den = 0.f;
    for (int j = 0; j < M_; ++j) den += __expf(logits[lane * M_ + j] - mx);
    lr_ws[(size_t)(b * T_ + t0 + lane) * M_ + m] =
        __expf(logits[lane * M_ + m] - mx) / den;
  }
  wait_async0();
}

// ---- kernel 2: feed GEMM -> register-resident scan -> complex readout -----
// grid: B*M blocks (independent recurrences), 256 threads = 8 waves.

__global__ __launch_bounds__(256) void k2_scan(
    const _Float16* __restrict__ xp_ws, const float* __restrict__ lr_ws,
    const _Float16* __restrict__ winReT, const _Float16* __restrict__ winImT,
    const _Float16* __restrict__ woutReT, const _Float16* __restrict__ woutImNegT,
    const float2* __restrict__ Lambda, float* __restrict__ out)
{
  __shared__ _Float16 xpTile[16 * D_];     // 2 KB
  __shared__ float    lrT[16];
  __shared__ float    feedRe[16 * R_];     // 8 KB
  __shared__ float    feedIm[16 * R_];     // 8 KB
  __shared__ _Float16 sRe16[16 * R_];      // 4 KB
  __shared__ _Float16 sIm16[16 * R_];      // 4 KB
  __shared__ _Float16 wore[O_ * R_];       // 8 KB
  __shared__ _Float16 woin[O_ * R_];       // 8 KB

  int tid = threadIdx.x;
  int b = blockIdx.x / M_, m = blockIdx.x % M_;
  int w = tid >> 5, lane = tid & 31;
  int h = lane >> 4, nl = lane & 15;

  // DMA the (tiny, reused 128x) readout weights straight into LDS
  for (int i = tid; i < O_ * R_ / 8; i += 256) {   // 512 x 16B segments each
    async_load_b128(lds_addr(&wore[i * 8]),
                    woutReT + (size_t)m * O_ * R_ + i * 8);
    async_load_b128(lds_addr(&woin[i * 8]),
                    woutImNegT + (size_t)m * O_ * R_ + i * 8);
  }

  const _Float16* winRe = winReT + (size_t)m * R_ * D_;
  const _Float16* winIm = winImT + (size_t)m * R_ * D_;

  float sRe = 0.f, sIm = 0.f, lamRe = 0.f, lamIm = 0.f;
  if (tid < R_) { float2 l = Lambda[m * R_ + tid]; lamRe = l.x; lamIm = l.y; }
  wait_async0();
  __syncthreads();

  for (int tile = 0; tile < T_ / 16; ++tile) {
    int t0 = tile * 16;

    // async DMA of the X_proj tile (128 lanes x 16B -> 2 KB LDS)
    if (tid < 128) {
      int row = tid >> 3, seg = tid & 7;
      const _Float16* g =
          xp_ws + ((size_t)(b * T_ + t0 + row) * M_ + m) * D_ + seg * 8;
      async_load_b128(lds_addr(&xpTile[row * D_ + seg * 8]), g);
    }
    if (tid < 16) lrT[tid] = lr_ws[(size_t)(b * T_ + t0 + tid) * M_ + m];
    if (tile + 1 < T_ / 16)   // gfx1250 global_prefetch_b8 for the next tile
      __builtin_prefetch(&xp_ws[((size_t)(b * T_ + t0 + 16) * M_ + m) * D_], 0, 1);
    wait_async0();
    __syncthreads();

    { // feed = X_proj @ Win (Re & Im); wave w owns r-tile w
      v16h a0 = load_a_frag(xpTile, D_, lane, 0);
      v16h a1 = load_a_frag(xpTile, D_, lane, 32);
      v8f accRe = {}, accIm = {};
      accRe = wmma_f16(a0, load_b_frag(winRe, D_, lane, w * 16, 0),  accRe);
      accRe = wmma_f16(a1, load_b_frag(winRe, D_, lane, w * 16, 32), accRe);
      accIm = wmma_f16(a0, load_b_frag(winIm, D_, lane, w * 16, 0),  accIm);
      accIm = wmma_f16(a1, load_b_frag(winIm, D_, lane, w * 16, 32), accIm);
#pragma unroll
      for (int g = 0; g < 8; ++g) {
        int t = g + 8 * h, r = w * 16 + nl;
        feedRe[t * R_ + r] = accRe[g];
        feedIm[t * R_ + r] = accIm[g];
      }
    }
    __syncthreads();

    // sequential 16-step complex scan; state lives in registers across all T
    if (tid < R_) {
      int r = tid;
#pragma unroll
      for (int tt = 0; tt < 16; ++tt) {
        float lr = lrT[tt];
        float LR = lr * lamRe + (1.f - lr);   // eff. lambda = lr*Lam + (1-lr)
        float LI = lr * lamIm;
        float fR = lr * feedRe[tt * R_ + r];  // feed scaled by lr
        float fI = lr * feedIm[tt * R_ + r];
        float nR = LR * sRe - LI * sIm + fR;
        float nI = LR * sIm + LI * sRe + fI;
        sRe = nR; sIm = nI;
        sRe16[tt * R_ + r] = (_Float16)sRe;
        sIm16[tt * R_ + r] = (_Float16)sIm;
      }
    }
    __syncthreads();

    // readout: Re(s @ Wout) = sRe@WoutRe + sIm@(-WoutIm); waves 0,1 = o-tiles
    if (w < 2) {
      v8f acc = {};
#pragma unroll
      for (int kk = 0; kk < 4; ++kk) {
        v16h ar = load_a_frag(sRe16, R_, lane, kk * 32);
        v16h ai = load_a_frag(sIm16, R_, lane, kk * 32);
        acc = wmma_f16(ar, load_b_frag(wore, R_, lane, w * 16, kk * 32), acc);
        acc = wmma_f16(ai, load_b_frag(woin, R_, lane, w * 16, kk * 32), acc);
      }
#pragma unroll
      for (int g = 0; g < 8; ++g) {
        int t = t0 + g + 8 * h, o = w * 16 + nl;
        out[((size_t)(b * T_ + t) * M_ + m) * O_ + o] = acc[g];
      }
    }
    __syncthreads();
  }
}

// ---- launcher -------------------------------------------------------------

extern "C" void kernel_launch(void* const* d_in, const int* in_sizes, int n_in,
                              void* d_out, int out_size, void* d_ws, size_t ws_size,
                              hipStream_t stream) {
  const float*  X    = (const float*)d_in[0];
  const float*  proj = (const float*)d_in[1];
  const float*  alr  = (const float*)d_in[2];
  const float*  temp = (const float*)d_in[3];
  const float2* Win  = (const float2*)d_in[4];   // complex64
  const float2* Lam  = (const float2*)d_in[5];   // complex64
  const float2* Wout = (const float2*)d_in[6];   // complex64
  float* out = (float*)d_out;

  char* ws = (char*)d_ws;
  size_t off = 0;
  _Float16* xp_ws      = (_Float16*)(ws + off); off += (size_t)B_ * T_ * M_ * D_ * 2;
  float*    lr_ws      = (float*)(ws + off);    off += (size_t)B_ * T_ * M_ * 4;
  _Float16* projT      = (_Float16*)(ws + off); off += (size_t)M_ * D_ * D_ * 2;
  _Float16* winReT     = (_Float16*)(ws + off); off += (size_t)M_ * D_ * R_ * 2;
  _Float16* winImT     = (_Float16*)(ws + off); off += (size_t)M_ * D_ * R_ * 2;
  _Float16* woutReT    = (_Float16*)(ws + off); off += (size_t)M_ * O_ * R_ * 2;
  _Float16* woutImNegT = (_Float16*)(ws + off); off += (size_t)M_ * O_ * R_ * 2;
  (void)ws_size; (void)in_sizes; (void)n_in; (void)out_size;

  k0_prep<<<256, 256, 0, stream>>>(proj, Win, Wout, projT, winReT, winImT,
                                   woutReT, woutImNegT);
  k1_proj_lr<<<B_ * (T_ / 16), 256, 0, stream>>>(X, projT, alr, temp, xp_ws, lr_ws);
  k2_scan<<<B_ * M_, 256, 0, stream>>>(xp_ws, lr_ws, winReT, winImT,
                                       woutReT, woutImNegT, Lam, out);
}
